// MultiHeadAttention_30331059044503
// MI455X (gfx1250) — matrix operand order repaired and checked by reading, not compiled, not run on hardware
//
#include <hip/hip_runtime.h>

typedef __attribute__((ext_vector_type(16))) __bf16 v16bf;
typedef __attribute__((ext_vector_type(8)))  __bf16 v8bf;
typedef __attribute__((ext_vector_type(4)))  __bf16 v4bf;
typedef __attribute__((ext_vector_type(8)))  float  v8f;
typedef __attribute__((ext_vector_type(8)))  float  v8ff;

constexpr int B_  = 4;
constexpr int S_  = 2048;
constexpr int D_  = 1024;
constexpr int H_  = 16;
constexpr int DK_ = 64;
constexpr int PLD = 40;  // attn P-staging LDS row stride (bf16): 80B, 16B-aligned
constexpr int TLD = 36;  // proj V-transpose LDS row stride (bf16): 72B, 8B-aligned

__device__ __forceinline__ v8f vzero8() {
  v8f z = {0.f, 0.f, 0.f, 0.f, 0.f, 0.f, 0.f, 0.f};
  return z;
}

__device__ __forceinline__ v8f wmma_bf16(v16bf a, v16bf b, v8f c) {
  // D = A(16x32 bf16) * B(32x16 bf16) + C(16x16 f32)
  return __builtin_amdgcn_wmma_f32_16x16x32_bf16(false, a, false, b, (short)0, c,
                                                 false, false);
}

// ---- Fragment loaders ----------------------------------------------------
// A 16x32 bf16: lanes 0-15: elems j -> K j; lanes 16-31: elems j -> K 16+j
//               (same contiguous k order as b_frag_bf16)
// rowptr already offset to this lane's row (m = lane & 15).
__device__ __forceinline__ v16bf a_frag_bf16(const __bf16* rowptr, int kbase, int half) {
  const __bf16* p0 = rowptr + kbase + 16 * half;
  v8bf lo = *(const v8bf*)p0;
  v8bf hi = *(const v8bf*)(p0 + 8);
  return __builtin_shufflevector(lo, hi, 0, 1, 2, 3, 4, 5, 6, 7,
                                 8, 9, 10, 11, 12, 13, 14, 15);
}

// B 32x16 bf16: lanes 0-15: elems j -> K j; lanes 16-31: elems j -> K 16+j.
// colptr points at the contiguous K-run for this lane's column (n = lane & 15).
__device__ __forceinline__ v16bf b_frag_bf16(const __bf16* colptr, int kbase, int half) {
  return *(const v16bf*)(colptr + kbase + 16 * half);
}

// ---- Stage 0: fp32 -> bf16 conversion (8 elems/thread) ------------------
__global__ __launch_bounds__(256) void cvt_kernel(const float* __restrict__ src,
                                                  __bf16* __restrict__ dst) {
  const size_t i = (size_t)(blockIdx.x * blockDim.x + threadIdx.x) * 8;
  v8ff f = *(const v8ff*)(src + i);
  v8bf o;
#pragma unroll
  for (int j = 0; j < 8; ++j) o[j] = (__bf16)f[j];
  *(v8bf*)(dst + i) = o;
}

// ---- Stage 1: Y = X * W^T + b (all bf16 operands, f32 accum) ------------
// Wave tile: 32 rows x 64 cols (one head's worth of columns).
// mode 0: dst [B,H,S,DK].  mode 2: dst [B,H,DK,S], coalesced via LDS transpose.
__global__ __launch_bounds__(256) void proj_kernel(
    const __bf16* __restrict__ X, const __bf16* __restrict__ W,
    const float* __restrict__ bias, __bf16* __restrict__ dst, int mode) {
  extern __shared__ __bf16 tstage[];  // mode 2 only: 8 waves * 64*TLD bf16
  const int wave = (blockIdx.x * blockDim.x + threadIdx.x) >> 5;
  const int wib  = threadIdx.x >> 5;
  const int lane = threadIdx.x & 31;
  const int half = lane >> 4;
  const int mrow = lane & 15;
  const int tileM = wave >> 4;         // 256 row-tiles of 32
  const int n0    = (wave & 15) * 64;  // 16 col-groups of 64 (== one head)

  const __bf16* arow0 = X + (size_t)(tileM * 32 + mrow) * D_;
  const __bf16* arow1 = arow0 + 16 * D_;
  const __bf16* wcol0 = W + (size_t)(n0 + mrow) * D_;

  v8f acc[2][4];
#pragma unroll
  for (int s = 0; s < 2; ++s)
#pragma unroll
    for (int t = 0; t < 4; ++t) acc[s][t] = vzero8();

  for (int kk = 0; kk < D_; kk += 32) {
    v16bf a0 = a_frag_bf16(arow0, kk, half);
    v16bf a1 = a_frag_bf16(arow1, kk, half);
    v16bf b0 = b_frag_bf16(wcol0, kk, half);
    v16bf b1 = b_frag_bf16(wcol0 + 16 * D_, kk, half);
    v16bf b2 = b_frag_bf16(wcol0 + 32 * D_, kk, half);
    v16bf b3 = b_frag_bf16(wcol0 + 48 * D_, kk, half);
    acc[0][0] = wmma_bf16(a0, b0, acc[0][0]);
    acc[1][0] = wmma_bf16(a1, b0, acc[1][0]);
    acc[0][1] = wmma_bf16(a0, b1, acc[0][1]);
    acc[1][1] = wmma_bf16(a1, b1, acc[1][1]);
    acc[0][2] = wmma_bf16(a0, b2, acc[0][2]);
    acc[1][2] = wmma_bf16(a1, b2, acc[1][2]);
    acc[0][3] = wmma_bf16(a0, b3, acc[0][3]);
    acc[1][3] = wmma_bf16(a1, b3, acc[1][3]);
  }

  if (mode == 0) {
#pragma unroll
    for (int t = 0; t < 4; ++t) {
      const int n  = n0 + t * 16 + mrow;
      const float bv = bias[n];
      const int h = n >> 6;
      const int d = n & 63;
#pragma unroll
      for (int s2 = 0; s2 < 2; ++s2) {
#pragma unroll
        for (int r = 0; r < 8; ++r) {
          const int row = tileM * 32 + s2 * 16 + r + 8 * half;
          const int bb  = row >> 11;
          const int s   = row & (S_ - 1);
          dst[((size_t)(bb * H_ + h) * S_ + s) * DK_ + d] =
              (__bf16)(acc[s2][t][r] + bv);
        }
      }
    }
  } else {
    // Stage tile as [d][s] in LDS (scattered 2B LDS writes are cheap), then
    // emit coalesced global b64 stores along s.
    __bf16* tb = tstage + wib * 64 * TLD;
#pragma unroll
    for (int t = 0; t < 4; ++t) {
      const float bv = bias[n0 + t * 16 + mrow];
      const int d = t * 16 + mrow;  // 64-col group == one head, so d = n & 63
#pragma unroll
      for (int s2 = 0; s2 < 2; ++s2) {
#pragma unroll
        for (int r = 0; r < 8; ++r) {
          const int srow = s2 * 16 + r + 8 * half;  // 0..31 within tile
          tb[d * TLD + srow] = (__bf16)(acc[s2][t][r] + bv);
        }
      }
    }
    asm volatile("s_wait_dscnt 0x0" ::: "memory");

    const int row0  = tileM * 32;
    const int bb    = row0 >> 11;
    const int sbase = row0 & (S_ - 1);
    const int h     = n0 >> 6;
    __bf16* vdst = dst + ((size_t)(bb * H_ + h) * DK_) * S_ + sbase;
    const int dl = lane >> 3;        // 0..3
    const int so = (lane & 7) * 4;   // 0,4,..,28
#pragma unroll
    for (int i = 0; i < 16; ++i) {
      const int d = i * 4 + dl;
      v4bf v = *(const v4bf*)(tb + d * TLD + so);
      *(v4bf*)(vdst + (size_t)d * S_ + so) = v;
    }
  }
}

// ---- Stage 2: flash attention per (b,h), one wave per 16-query block ----
__global__ __launch_bounds__(256) void attn_kernel(
    const __bf16* __restrict__ Q, const __bf16* __restrict__ K,
    const __bf16* __restrict__ Vt, __bf16* __restrict__ ctx) {
  __shared__ __bf16 lds[8 * 16 * PLD];
  const int wave = (blockIdx.x * blockDim.x + threadIdx.x) >> 5;
  const int wib  = threadIdx.x >> 5;
  __bf16* pbuf = lds + wib * 16 * PLD;
  const int lane = threadIdx.x & 31;
  const int half = lane >> 4;
  const int mrow = lane & 15;

  const int qblk = wave & 127;  // S/16
  const int bh   = wave >> 7;   // b*H + h
  const int h    = bh & (H_ - 1);
  const int b    = bh >> 4;

  const __bf16* Qp = Q  + ((size_t)bh * S_ + (size_t)qblk * 16) * DK_;
  const __bf16* Kp = K  + (size_t)bh * S_ * DK_;
  const __bf16* Vp = Vt + (size_t)bh * DK_ * S_;

  v16bf qa0 = a_frag_bf16(Qp + (size_t)mrow * DK_, 0, half);
  v16bf qa1 = a_frag_bf16(Qp + (size_t)mrow * DK_, 32, half);

  float rm[8], rl[8];
  v8f o[4];
#pragma unroll
  for (int r = 0; r < 8; ++r) { rm[r] = -3.0e30f; rl[r] = 0.f; }
#pragma unroll
  for (int t = 0; t < 4; ++t) o[t] = vzero8();

  for (int kc = 0; kc < S_; kc += 32) {
    // Scores: S = Q * K^T / 8. B[k=d][n=key]: row (kc+n) of K, contiguous in d.
    const __bf16* kcol0 = Kp + (size_t)(kc + mrow) * DK_;
    const __bf16* kcol1 = kcol0 + 16 * DK_;
    v8f s0 = vzero8(), s1 = vzero8();
    s0 = wmma_bf16(qa0, b_frag_bf16(kcol0, 0, half), s0);
    s0 = wmma_bf16(qa1, b_frag_bf16(kcol0, 32, half), s0);
    s1 = wmma_bf16(qa0, b_frag_bf16(kcol1, 0, half), s1);
    s1 = wmma_bf16(qa1, b_frag_bf16(kcol1, 32, half), s1);

    // Issue V fragments now so the loads overlap the softmax VALU chain.
    v16bf vb[4];
#pragma unroll
    for (int t = 0; t < 4; ++t)
      vb[t] = b_frag_bf16(Vp + (size_t)(t * 16 + mrow) * S_ + kc, 0, half);

    // Online softmax: row m's 16 cols live in one VGPR across a 16-lane half.
#pragma unroll
    for (int r = 0; r < 8; ++r) {
      const float v0 = s0[r] * 0.125f;  // 1/sqrt(64)
      const float v1 = s1[r] * 0.125f;
      float mx = fmaxf(v0, v1);
      mx = fmaxf(mx, __shfl_xor(mx, 1));
      mx = fmaxf(mx, __shfl_xor(mx, 2));
      mx = fmaxf(mx, __shfl_xor(mx, 4));
      mx = fmaxf(mx, __shfl_xor(mx, 8));
      const float mnew = fmaxf(rm[r], mx);
      const float p0 = __expf(v0 - mnew);
      const float p1 = __expf(v1 - mnew);
      float sum = p0 + p1;
      sum += __shfl_xor(sum, 1);
      sum += __shfl_xor(sum, 2);
      sum += __shfl_xor(sum, 4);
      sum += __shfl_xor(sum, 8);
      const float cs = __expf(rm[r] - mnew);
      rl[r] = rl[r] * cs + sum;
      rm[r] = mnew;
      const int row = r + 8 * half;
      pbuf[row * PLD + mrow]      = (__bf16)p0;  // keys kc + 0..15
      pbuf[row * PLD + 16 + mrow] = (__bf16)p1;  // keys kc + 16..31
#pragma unroll
      for (int t = 0; t < 4; ++t) o[t][r] *= cs;
    }
    asm volatile("s_wait_dscnt 0x0" ::: "memory");

    // Re-read P (16x32) as an A fragment; LDS row stride PLD bf16.
    v16bf pa = a_frag_bf16(pbuf + mrow * PLD, 0, half);

#pragma unroll
    for (int t = 0; t < 4; ++t) o[t] = wmma_bf16(pa, vb[t], o[t]);
  }

  // Normalize and scatter ctx (bf16, [B,S,D]).
#pragma unroll
  for (int t = 0; t < 4; ++t) {
#pragma unroll
    for (int r = 0; r < 8; ++r) {
      const int qrow = qblk * 16 + r + 8 * half;
      const float val = o[t][r] / rl[r];
      ctx[((size_t)(b * S_ + qrow)) * D_ + h * DK_ + t * 16 + mrow] = (__bf16)val;
    }
  }
}

// ---- Stage 3: out = ctx * Wo^T + bo, fp32 out. 32x64 wave tile ----------
__global__ __launch_bounds__(256) void out_proj_kernel(
    const __bf16* __restrict__ ctx, const __bf16* __restrict__ W,
    const float* __restrict__ bias, float* __restrict__ out) {
  const int wave = (blockIdx.x * blockDim.x + threadIdx.x) >> 5;
  const int lane = threadIdx.x & 31;
  const int half = lane >> 4;
  const int mrow = lane & 15;
  const int tileM = wave >> 4;
  const int n0    = (wave & 15) * 64;

  const __bf16* arow0 = ctx + (size_t)(tileM * 32 + mrow) * D_;
  const __bf16* arow1 = arow0 + 16 * D_;
  const __bf16* wcol0 = W + (size_t)(n0 + mrow) * D_;

  v8f acc[2][4];
#pragma unroll
  for (int s = 0; s < 2; ++s)
#pragma unroll
    for (int t = 0; t < 4; ++t) acc[s][t] = vzero8();

  for (int kk = 0; kk < D_; kk += 32) {
    v16bf a0 = a_frag_bf16(arow0, kk, half);
    v16bf a1 = a_frag_bf16(arow1, kk, half);
    v16bf b0 = b_frag_bf16(wcol0, kk, half);
    v16bf b1 = b_frag_bf16(wcol0 + 16 * D_, kk, half);
    v16bf b2 = b_frag_bf16(wcol0 + 32 * D_, kk, half);
    v16bf b3 = b_frag_bf16(wcol0 + 48 * D_, kk, half);
    acc[0][0] = wmma_bf16(a0, b0, acc[0][0]);
    acc[1][0] = wmma_bf16(a1, b0, acc[1][0]);
    acc[0][1] = wmma_bf16(a0, b1, acc[0][1]);
    acc[1][1] = wmma_bf16(a1, b1, acc[1][1]);
    acc[0][2] = wmma_bf16(a0, b2, acc[0][2]);
    acc[1][2] = wmma_bf16(a1, b2, acc[1][2]);
    acc[0][3] = wmma_bf16(a0, b3, acc[0][3]);
    acc[1][3] = wmma_bf16(a1, b3, acc[1][3]);
  }

#pragma unroll
  for (int t = 0; t < 4; ++t) {
    const int n = n0 + t * 16 + mrow;
    const float bv = bias[n];
#pragma unroll
    for (int s2 = 0; s2 < 2; ++s2) {
#pragma unroll
      for (int r = 0; r < 8; ++r) {
        const int row = tileM * 32 + s2 * 16 + r + 8 * half;
        out[(size_t)row * D_ + n] = acc[s2][t][r] + bv;
      }
    }
  }
}

extern "C" void kernel_launch(void* const* d_in, const int* in_sizes, int n_in,
                              void* d_out, int out_size, void* d_ws, size_t ws_size,
                              hipStream_t stream) {
  const float* query = (const float*)d_in[0];
  const float* key   = (const float*)d_in[1];
  const float* value = (const float*)d_in[2];
  const float* Wq = (const float*)d_in[3];
  const float* bq = (const float*)d_in[4];
  const float* Wk = (const float*)d_in[5];
  const float* bk = (const float*)d_in[6];
  const float* Wv = (const float*)d_in[7];
  const float* bv = (const float*)d_in[8];
  const float* Wo = (const float*)d_in[9];
  const float* bo = (const float*)d_in[10];

  const size_t act_elems = (size_t)B_ * S_ * D_;  // 8.39M
  const size_t w_elems   = (size_t)D_ * D_;       // 1.05M

  __bf16* Wqb = (__bf16*)d_ws;
  __bf16* Wkb = Wqb + w_elems;
  __bf16* Wvb = Wkb + w_elems;
  __bf16* Wob = Wvb + w_elems;
  __bf16* Qws = Wob + w_elems;
  __bf16* Kws = Qws + act_elems;
  __bf16* Vws = Kws + act_elems;   // transposed [B,H,DK,S]
  __bf16* Xb  = Vws + act_elems;   // bf16 activation staging
  __bf16* Cws = Xb;                // ctx aliases Xb (dead after V projection)

  const dim3 blk(256);
  const dim3 cvtW_grid(w_elems / 8 / 256);            // 512
  const dim3 cvtX_grid(act_elems / 8 / 256);          // 4096
  const int gemm_waves = (B_ * S_ / 32) * (D_ / 64);  // 4096
  const dim3 gemm_grid(gemm_waves / 8);               // 512
  const int attn_waves = B_ * H_ * (S_ / 16);         // 8192
  const dim3 attn_grid(attn_waves / 8);               // 1024
  const size_t tr_shmem = 8 * 64 * TLD * sizeof(__bf16);  // 36.9KB, mode 2 only

  cvt_kernel<<<cvtW_grid, blk, 0, stream>>>(Wq, Wqb);
  cvt_kernel<<<cvtW_grid, blk, 0, stream>>>(Wk, Wkb);
  cvt_kernel<<<cvtW_grid, blk, 0, stream>>>(Wv, Wvb);
  cvt_kernel<<<cvtW_grid, blk, 0, stream>>>(Wo, Wob);

  cvt_kernel<<<cvtX_grid, blk, 0, stream>>>(query, Xb);
  proj_kernel<<<gemm_grid, blk, 0, stream>>>(Xb, Wqb, bq, Qws, 0);
  cvt_kernel<<<cvtX_grid, blk, 0, stream>>>(key, Xb);
  proj_kernel<<<gemm_grid, blk, 0, stream>>>(Xb, Wkb, bk, Kws, 0);
  cvt_kernel<<<cvtX_grid, blk, 0, stream>>>(value, Xb);
  proj_kernel<<<gemm_grid, blk, tr_shmem, stream>>>(Xb, Wvb, bv, Vws, 2);

  attn_kernel<<<attn_grid, blk, 0, stream>>>(Qws, Kws, Vws, Cws);
  out_proj_kernel<<<gemm_grid, blk, 0, stream>>>(Cws, Wob, bo, (float*)d_out);
}
